// MultiHeadAttention_43258910605919
// MI455X (gfx1250) — compile-verified, matrix operand order checked by reading, not run on hardware
//
#include <hip/hip_runtime.h>
#include <hip/hip_bf16.h>

// ---------------------------------------------------------------------------
// MultiHeadAttention with graph biases, CDNA5 (gfx1250) WMMA implementation.
// B=8, S=1024, D=512, H=8, depth=64.  All matmuls run on
// v_wmma_f32_16x16x32_bf16 (f32 accumulate).  attn tensor + adjoin/dist are
// touched exactly once -> kernel is HBM-bound (~0.9 GB @ 23.3 TB/s).
// attn output is streamed LDS->HBM with GLOBAL_STORE_ASYNC_FROM_LDS_B128,
// overlapped with the PV WMMA stage; dense-kernel A tiles are staged with
// GLOBAL_LOAD_ASYNC_TO_LDS_B128 (pure bf16 byte copy, no VGPR round trip).
// ---------------------------------------------------------------------------

#define BB 8
#define SS 1024
#define DD 512
#define HH 8
#define DEPTH 64

typedef __attribute__((ext_vector_type(16))) __bf16 v16bf;
typedef __attribute__((ext_vector_type(8)))  __bf16 v8bf;
typedef __attribute__((ext_vector_type(4)))  __bf16 v4bf;
typedef __attribute__((ext_vector_type(8)))  float  v8f;

// ---- WMMA helper ----------------------------------------------------------
__device__ __forceinline__ v8f wmma_bf16(v16bf a, v16bf b, v8f c) {
  // (neg_a, A, neg_b, B, c_mod, C, reuse_a, reuse_b)
  return __builtin_amdgcn_wmma_f32_16x16x32_bf16(false, a, false, b,
                                                 (short)0, c, false, false);
}

// ---- CDNA5 async LDS<->global ops (ASYNCcnt-tracked), inline asm ----------
// Per-lane 16B transfers; LDS byte address = low 32 bits of generic pointer
__device__ __forceinline__ void async_ld_b128(unsigned lds_off,
                                              const void* gaddr) {
  asm volatile("global_load_async_to_lds_b128 %0, %1, off"
               :: "v"(lds_off), "v"(gaddr) : "memory");
}
__device__ __forceinline__ void async_st_b128(void* gaddr, unsigned lds_off) {
  asm volatile("global_store_async_from_lds_b128 %0, %1, off"
               :: "v"(gaddr), "v"(lds_off) : "memory");
}
__device__ __forceinline__ void wait_async0() {
  asm volatile("s_wait_asynccnt 0x0" ::: "memory");
}

// ---- 16-bit A/B fragment load (ISA 7.12.2 layout) -------------------------
// lanes 0-15 : row = lane,    K = 0..7  (v0..v3), K = 16..23 (v4..v7)
// lanes 16-31: row = lane-16, K = 8..15,          K = 24..31
// => per lane: two contiguous 8-element (16B) loads at K-offsets k0 and k0+16.
__device__ __forceinline__ v16bf frag_ld(const __bf16* base, int ld) {
  int lane = threadIdx.x & 31;
  int row  = lane & 15;
  int k0   = (lane >> 4) << 3;            // 0 or 8
  const __bf16* p = base + row * ld + k0;
  v8bf lo = *(const v8bf*)(p);
  v8bf hi = *(const v8bf*)(p + 16);
  return __builtin_shufflevector(lo, hi, 0, 1, 2, 3, 4, 5, 6, 7,
                                         8, 9, 10, 11, 12, 13, 14, 15);
}

__device__ __forceinline__ v4bf cvt4(float4 f) {
  v4bf h;
  h[0] = (__bf16)f.x; h[1] = (__bf16)f.y;
  h[2] = (__bf16)f.z; h[3] = (__bf16)f.w;
  return h;
}

__device__ __forceinline__ v8f vzero8() {
  v8f z = {0.f, 0.f, 0.f, 0.f, 0.f, 0.f, 0.f, 0.f};
  return z;
}

// ===========================================================================
// Kernel 1: QKV projection.  out = X @ W^T + bias   (M=8192, N=512, K=512)
// blockIdx.z selects q/k/v.  Stores bf16: qh,kh -> [B,H,S,64],
// vh -> transposed [B,H,64,S] so PV-stage B-fragments are contiguous.
// ===========================================================================
__global__ __launch_bounds__(256)
void qkv_proj_kernel(const float* __restrict__ q, const float* __restrict__ k,
                     const float* __restrict__ v,
                     const float* __restrict__ wq, const float* __restrict__ bq,
                     const float* __restrict__ wk, const float* __restrict__ bk,
                     const float* __restrict__ wv, const float* __restrict__ bv,
                     __bf16* __restrict__ qh, __bf16* __restrict__ kh,
                     __bf16* __restrict__ vhT) {
  __shared__ __bf16 As[128][40];   // 128x32 bf16 tile (+pad)
  __shared__ __bf16 Bs[128][40];

  const int z = blockIdx.z;
  const float* X    = (z == 0) ? q  : (z == 1) ? k  : v;
  const float* W    = (z == 0) ? wq : (z == 1) ? wk : wv;
  const float* bias = (z == 0) ? bq : (z == 1) ? bk : bv;

  const int m0 = blockIdx.x * 128;
  const int n0 = blockIdx.y * 128;
  const int tid = threadIdx.x, lane = tid & 31, wave = tid >> 5;
  const int wm = (wave & 3) * 32;     // wave's row block inside tile
  const int wn = (wave >> 2) * 64;    // wave's col block inside tile

  v8f acc[2][4];
  for (int mt = 0; mt < 2; ++mt)
    for (int nt = 0; nt < 4; ++nt) acc[mt][nt] = vzero8();

  for (int kc = 0; kc < DD; kc += 32) {
    __syncthreads();
    // stage A (X rows) and B (W rows, since out = X*W^T) into LDS as bf16
    for (int i = tid; i < 1024; i += 256) {        // 1024 float4 per tile
      int r = i >> 3, c = (i & 7) * 4;
      float4 fa = *(const float4*)(X + (long)(m0 + r) * DD + kc + c);
      float4 fb = *(const float4*)(W + (long)(n0 + r) * DD + kc + c);
      *(v4bf*)&As[r][c] = cvt4(fa);
      *(v4bf*)&Bs[r][c] = cvt4(fb);
      if (kc + 32 < DD) {                          // global_prefetch_b8
        __builtin_prefetch(X + (long)(m0 + r) * DD + kc + 32 + c, 0, 1);
        __builtin_prefetch(W + (long)(n0 + r) * DD + kc + 32 + c, 0, 1);
      }
    }
    __syncthreads();

    v16bf af[2], bf_[4];
    af[0] = frag_ld(&As[wm][0], 40);
    af[1] = frag_ld(&As[wm + 16][0], 40);
#pragma unroll
    for (int nt = 0; nt < 4; ++nt) bf_[nt] = frag_ld(&Bs[wn + nt * 16][0], 40);
#pragma unroll
    for (int mt = 0; mt < 2; ++mt)
#pragma unroll
      for (int nt = 0; nt < 4; ++nt)
        acc[mt][nt] = wmma_bf16(af[mt], bf_[nt], acc[mt][nt]);
  }

  // C/D layout: element (r, lane): M = r + 8*(lane>=16), N = lane&15
#pragma unroll
  for (int mt = 0; mt < 2; ++mt)
#pragma unroll
    for (int nt = 0; nt < 4; ++nt)
#pragma unroll
      for (int r = 0; r < 8; ++r) {
        int gm = m0 + wm + mt * 16 + r + ((lane >> 4) << 3);
        int gn = n0 + wn + nt * 16 + (lane & 15);
        __bf16 val = (__bf16)(acc[mt][nt][r] + bias[gn]);
        int bb = gm >> 10, s = gm & 1023;           // gm = b*S + s
        int h  = gn >> 6,  d = gn & 63;             // gn = h*64 + d
        if (z == 0)
          qh[((long)(bb * HH + h) * SS + s) * DEPTH + d] = val;
        else if (z == 1)
          kh[((long)(bb * HH + h) * SS + s) * DEPTH + d] = val;
        else
          vhT[((long)(bb * HH + h) * DEPTH + d) * SS + s] = val;
      }
}

// ===========================================================================
// Kernel 2: attention.  One WG per (b, h, 32 query rows).
//   phase 1: logits = qh*kh^T/8 (+mask,+adjoin,+dist) -> 32x1024 f32 in LDS
//   phase 2: row softmax (shfl_xor reductions); normalized probs written back
//            into the LDS tile (same ordering as global attn block) + bf16 P
//   phase 2b: 128 KB LDS tile -> global attn via async B128 stores,
//             overlapped with phase 3
//   phase 3: ctx = P @ vh via WMMA (vh stored transposed)
// Dynamic LDS: 32*1024*4 + 32*1024*2 = 192 KB (<= 320 KB/WG on CDNA5).
// ===========================================================================
__global__ __launch_bounds__(256)
void attn_kernel(const __bf16* __restrict__ qh, const __bf16* __restrict__ kh,
                 const __bf16* __restrict__ vhT, const int* __restrict__ mask,
                 const float* __restrict__ adjoin,
                 const float* __restrict__ dist,
                 float* __restrict__ attn_out, __bf16* __restrict__ ctx) {
  extern __shared__ char smem[];
  float*  Lg = (float*)smem;                       // [32][1024] logits/probs
  __bf16* P  = (__bf16*)(smem + 32 * 1024 * 4);    // [32][1024] probs (bf16)

  const int qt = blockIdx.x, h = blockIdx.y, b = blockIdx.z;
  const int q0 = qt * 32;
  const __bf16* qh_bh = qh  + (long)(b * HH + h) * SS * DEPTH;
  const __bf16* kh_bh = kh  + (long)(b * HH + h) * SS * DEPTH;
  const __bf16* vT_bh = vhT + (long)(b * HH + h) * DEPTH * SS;

  const int tid = threadIdx.x, lane = tid & 31, wave = tid >> 5;
  const int mh = wave & 1;            // which 16 query rows
  const int ng = wave >> 1;           // key-tile group (4 groups x 256 keys)

  // ---- phase 1: logits -----------------------------------------------------
  v16bf a0 = frag_ld(qh_bh + (q0 + mh * 16) * DEPTH, DEPTH);
  v16bf a1 = frag_ld(qh_bh + (q0 + mh * 16) * DEPTH + 32, DEPTH);

  for (int nt = ng * 16; nt < ng * 16 + 16; ++nt) {
    int key0 = nt * 16;
    v16bf b0 = frag_ld(kh_bh + key0 * DEPTH, DEPTH);
    v16bf b1 = frag_ld(kh_bh + key0 * DEPTH + 32, DEPTH);
    v8f acc = vzero8();
    acc = wmma_bf16(a0, b0, acc);
    acc = wmma_bf16(a1, b1, acc);
#pragma unroll
    for (int r = 0; r < 8; ++r) {
      int ml  = mh * 16 + r + ((lane >> 4) << 3);   // local q row 0..31
      int qg  = q0 + ml;
      int key = key0 + (lane & 15);
      float l;
      if (mask[b * SS + key] == 0) {
        l = -1e30f;                                 // masked_fill(-inf)
      } else {
        long bi = ((long)b * SS + qg) * SS + key;
        l = acc[r] * 0.125f + adjoin[bi] + dist[bi];
      }
      Lg[ml * SS + key] = l;
    }
  }
  __syncthreads();

  // ---- phase 2: softmax (probs written back into Lg, bf16 copy in P) ------
  {
    int row = tid >> 3, seg = tid & 7;              // 8 lanes per row
    float* Lr = Lg + row * SS + seg * 128;
    float m = -3.4e38f;
    for (int i = 0; i < 128; ++i) m = fmaxf(m, Lr[i]);
    m = fmaxf(m, __shfl_xor(m, 1));
    m = fmaxf(m, __shfl_xor(m, 2));
    m = fmaxf(m, __shfl_xor(m, 4));
    float s = 0.f;
    for (int i = 0; i < 128; ++i) {
      float e = __expf(Lr[i] - m);
      Lr[i] = e;
      s += e;
    }
    s += __shfl_xor(s, 1);
    s += __shfl_xor(s, 2);
    s += __shfl_xor(s, 4);
    float inv = 1.f / s;
    __bf16* Pr = P + row * SS + seg * 128;
    for (int i = 0; i < 128; ++i) {
      float p = Lr[i] * inv;
      Lr[i] = p;                                    // normalized, in place
      Pr[i] = (__bf16)p;                            // bf16 copy for PV matmul
    }
  }
  __syncthreads();

  // ---- phase 2b: stream 128 KB prob tile LDS -> global attn (async) -------
  // attn[b,h, q0..q0+31, 0..1023] is one contiguous 128 KB block whose
  // ordering matches Lg exactly.  8 waves x 32 instrs x (32 lanes x 16B).
  {
    char* gbase = (char*)(attn_out + ((long)(b * HH + h) * SS + q0) * SS);
    unsigned lbase = (unsigned)(uintptr_t)Lg;
#pragma unroll 4
    for (int j = 0; j < 32; ++j) {
      unsigned off = (((unsigned)wave * 32u + j) * 32u + lane) * 16u;
      async_st_b128(gbase + off, lbase + off);
    }
  }

  // ---- phase 3: ctx = P @ V (overlaps with outstanding async stores) ------
  const int nc = (wave >> 1) * 16;                  // depth-column tile
  v8f acc = vzero8();
  for (int kc = 0; kc < SS; kc += 32) {
    v16bf a  = frag_ld(P + (mh * 16) * SS + kc, SS);
    v16bf bv = frag_ld(vT_bh + nc * SS + kc, SS);   // rows of V^T = cols of V
    acc = wmma_bf16(a, bv, acc);
  }
#pragma unroll
  for (int r = 0; r < 8; ++r) {
    int ml = mh * 16 + r + ((lane >> 4) << 3);
    int qg = q0 + ml;
    int d  = nc + (lane & 15);
    ctx[((long)b * SS + qg) * DD + h * DEPTH + d] = (__bf16)acc[r];
  }
  wait_async0();                                    // drain ASYNCcnt
}

// ===========================================================================
// Kernel 3: output dense.  out = ctx @ dense_w^T + dense_b  (f32 out)
// A tile (already bf16) staged via GLOBAL_LOAD_ASYNC_TO_LDS_B128.
// ===========================================================================
__global__ __launch_bounds__(256)
void dense_kernel(const __bf16* __restrict__ ctx, const float* __restrict__ W,
                  const float* __restrict__ bias, float* __restrict__ out) {
  __shared__ __bf16 As[128][40];
  __shared__ __bf16 Bs[128][40];

  const int m0 = blockIdx.x * 128;
  const int n0 = blockIdx.y * 128;
  const int tid = threadIdx.x, lane = tid & 31, wave = tid >> 5;
  const int wm = (wave & 3) * 32;
  const int wn = (wave >> 2) * 64;

  v8f acc[2][4];
  for (int mt = 0; mt < 2; ++mt)
    for (int nt = 0; nt < 4; ++nt) acc[mt][nt] = vzero8();

  for (int kc = 0; kc < DD; kc += 32) {
    __syncthreads();
    // A tile: pure bf16 byte copy -> async direct-to-LDS (no VGPR data)
    for (int i = tid; i < 512; i += 256) {          // 512 x 16B chunks
      int r = i >> 2, c = (i & 3) * 8;
      async_ld_b128((unsigned)(uintptr_t)&As[r][c],
                    ctx + (long)(m0 + r) * DD + kc + c);
    }
    // B tile: f32 -> bf16 conversion on the register path
    for (int i = tid; i < 1024; i += 256) {
      int r = i >> 3, c = (i & 7) * 4;
      float4 fb = *(const float4*)(W + (long)(n0 + r) * DD + kc + c);
      *(v4bf*)&Bs[r][c] = cvt4(fb);
      if (kc + 32 < DD)
        __builtin_prefetch(W + (long)(n0 + r) * DD + kc + 32 + c, 0, 1);
    }
    wait_async0();                                  // own wave's asyncs done
    __syncthreads();                                // -> all waves' visible

    v16bf af[2], bf_[4];
    af[0] = frag_ld(&As[wm][0], 40);
    af[1] = frag_ld(&As[wm + 16][0], 40);
#pragma unroll
    for (int nt = 0; nt < 4; ++nt) bf_[nt] = frag_ld(&Bs[wn + nt * 16][0], 40);
#pragma unroll
    for (int mt = 0; mt < 2; ++mt)
#pragma unroll
      for (int nt = 0; nt < 4; ++nt)
        acc[mt][nt] = wmma_bf16(af[mt], bf_[nt], acc[mt][nt]);
  }

#pragma unroll
  for (int mt = 0; mt < 2; ++mt)
#pragma unroll
    for (int nt = 0; nt < 4; ++nt)
#pragma unroll
      for (int r = 0; r < 8; ++r) {
        int gm = m0 + wm + mt * 16 + r + ((lane >> 4) << 3);
        int gn = n0 + wn + nt * 16 + (lane & 15);
        out[(long)gm * DD + gn] = acc[mt][nt][r] + bias[gn];
      }
}

// ===========================================================================
extern "C" void kernel_launch(void* const* d_in, const int* in_sizes, int n_in,
                              void* d_out, int out_size, void* d_ws,
                              size_t ws_size, hipStream_t stream) {
  const float* q       = (const float*)d_in[0];
  const float* k       = (const float*)d_in[1];
  const float* v       = (const float*)d_in[2];
  const int*   mask    = (const int*)d_in[3];
  const float* adjoin  = (const float*)d_in[4];
  const float* dist    = (const float*)d_in[5];
  const float* wq_w    = (const float*)d_in[6];
  const float* wq_b    = (const float*)d_in[7];
  const float* wk_w    = (const float*)d_in[8];
  const float* wk_b    = (const float*)d_in[9];
  const float* wv_w    = (const float*)d_in[10];
  const float* wv_b    = (const float*)d_in[11];
  const float* dense_w = (const float*)d_in[12];
  const float* dense_b = (const float*)d_in[13];

  // workspace layout (bf16 elements): qh | kh | vhT | ctx, 4,194,304 each
  const long NE = (long)BB * HH * SS * DEPTH;       // 4,194,304
  __bf16* wsb = (__bf16*)d_ws;
  __bf16* qh  = wsb;
  __bf16* kh  = wsb + NE;
  __bf16* vhT = wsb + 2 * NE;
  __bf16* ctx = wsb + 3 * NE;

  float* out_f = (float*)d_out;                     // [B,S,D] f32
  float* attn  = out_f + (long)BB * SS * DD;        // [B,H,S,S] f32

  dim3 g1((BB * SS) / 128, DD / 128, 3);            // (64, 4, 3)
  qkv_proj_kernel<<<g1, 256, 0, stream>>>(q, k, v, wq_w, wq_b, wk_w, wk_b,
                                          wv_w, wv_b, qh, kh, vhT);

  dim3 g2(SS / 32, HH, BB);                         // (32, 8, 8)
  size_t smem = 32 * 1024 * 4 + 32 * 1024 * 2;      // 192 KB
  attn_kernel<<<g2, 256, smem, stream>>>(qh, kh, vhT, mask, adjoin, dist,
                                         attn, ctx);

  dim3 g3((BB * SS) / 128, DD / 128, 1);            // (64, 4, 1)
  dense_kernel<<<g3, 256, 0, stream>>>(ctx, dense_w, dense_b, out_f);
}
